// CrystalGraphConvNet_20014547599840
// MI455X (gfx1250) — matrix operand order repaired and checked by reading, not compiled, not run on hardware
//
#include <hip/hip_runtime.h>
#include <hip/hip_bf16.h>
#include <math.h>

typedef __attribute__((ext_vector_type(16))) _Float16 v16h;
typedef __attribute__((ext_vector_type(8)))  float    v8f;

#define NATOM   60000
#define MNBR    12
#define FDIM    64
#define NBF     41
#define OA      92
#define NCRYS   2048
#define OE      8
#define EDIM    16
#define HDIM    128
#define KREAL   169     // 2F + NB
#define KP      192     // padded K (6 chunks of 32)
#define APB     8       // atoms per block
#define ROWS    (APB * MNBR)   // 96 rows per block

#define BN_RS   0.9999950000374998f   // 1/sqrt(1+1e-5)

// branch-free softplus: log1p(exp(x)) == max(x,0) + log(1 + exp(-|x|))
// exp argument is <= 0 so no overflow path needed -> pure v_exp/v_log VALU.
__device__ __forceinline__ float softplus_f(float x) {
    return fmaxf(x, 0.0f) + __logf(1.0f + __expf(-fabsf(x)));
}
__device__ __forceinline__ float sigmoid_f(float x) {
    return __builtin_amdgcn_rcpf(1.0f + __expf(-x));   // v_rcp_f32, no div expansion
}

union V16U {
    v16h h;
    float4 f[2];
};

// ---------------- atom embedding: x = atom_fea @ W_embed + b ----------------
__global__ void embed_kernel(const float* __restrict__ atom_fea,
                             const float* __restrict__ W, const float* __restrict__ b,
                             float* __restrict__ x, _Float16* __restrict__ x16) {
    int i = blockIdx.x * blockDim.x + threadIdx.x;
    if (i >= NATOM * FDIM) return;
    int a = i >> 6, f = i & 63;
    float acc = b[f];
    const float* row = atom_fea + a * OA;
    #pragma unroll 4
    for (int k = 0; k < OA; ++k) acc = fmaf(row[k], W[k * FDIM + f], acc);
    x[i] = acc;
    x16[i] = (_Float16)acc;
}

// ---------------- extra embedding: softplus(bn(extra_fea @ W + b)) ----------------
__global__ void extra_kernel(const float* __restrict__ extra_fea,
                             const float* __restrict__ W, const float* __restrict__ b,
                             const float* __restrict__ g, const float* __restrict__ bb,
                             float* __restrict__ out) {
    int i = blockIdx.x * blockDim.x + threadIdx.x;
    if (i >= NCRYS * EDIM) return;
    int c = i / EDIM, e = i % EDIM;
    float acc = b[e];
    const float* row = extra_fea + c * OE;
    #pragma unroll
    for (int k = 0; k < OE; ++k) acc = fmaf(row[k], W[k * EDIM + e], acc);
    acc = acc * (g[e] * BN_RS) + bb[e];
    out[i] = softplus_f(acc);
}

// ---------------- conv weight -> f16, transposed [col][Kpad] ----------------
__global__ void prepw_kernel(const float* __restrict__ conv_W, _Float16* __restrict__ w16) {
    int i = blockIdx.x * blockDim.x + threadIdx.x;
    if (i >= 3 * 2 * FDIM * KP) return;
    int l = i / (2 * FDIM * KP);
    int r = i % (2 * FDIM * KP);
    int col = r / KP, k = r % KP;
    float v = (k < KREAL) ? conv_W[(l * KREAL + k) * (2 * FDIM) + col] : 0.0f;
    w16[i] = (_Float16)v;
}

// ---------------- zero pooling scratch ----------------
__global__ void zero_kernel(float* __restrict__ p, int n) {
    int i = blockIdx.x * blockDim.x + threadIdx.x;
    if (i < n) p[i] = 0.0f;
}

// ---------------- CGCNN conv layer (WMMA + async-to-LDS gather) ----------------
__global__ __launch_bounds__(128) void conv_kernel(
    const float* __restrict__ x_in,           // f32 residual input
    const _Float16* __restrict__ x16_in,      // f16 mirror for A staging
    float* __restrict__ x_out, _Float16* __restrict__ x16_out,
    const float* __restrict__ nbr_fea, const int* __restrict__ nbr_idx,
    const _Float16* __restrict__ wl,          // [128][192] f16, this layer
    const float* __restrict__ cb,             // conv bias [128]
    const float* __restrict__ g1, const float* __restrict__ b1,
    const float* __restrict__ g2, const float* __restrict__ b2) {
    __shared__ _Float16 Ab[ROWS * KP];        // 36 KB staged A rows (f16)
    __shared__ float nsum[APB * FDIM];        // per-atom neighbor sums
    __shared__ int   sIdx[ROWS];              // gathered neighbor indices

    const int tid = threadIdx.x;
    const int atom0 = blockIdx.x * APB;

    for (int i = tid; i < APB * FDIM; i += 128) nsum[i] = 0.0f;
    if (tid < ROWS) {
        sIdx[tid] = nbr_idx[(atom0 + tid / MNBR) * MNBR + (tid % MNBR)];
    }
    __syncthreads();

    // ---- async gather of self|nbr feature rows straight into LDS ----
    // 96 rows x 16 chunks of 16B (self cols 0..63, nbr cols 64..127); 128 lanes x 12.
    const uint32_t abase = (uint32_t)(uintptr_t)Ab;
    #pragma unroll
    for (int it = 0; it < (ROWS * 16) / 128; ++it) {
        const int i = tid + it * 128;
        const int r = i >> 4, p = i & 15;
        const int src_row = (p < 8) ? (atom0 + r / MNBR) : sIdx[r];
        const _Float16* src = x16_in + (size_t)src_row * FDIM + (p & 7) * 8;
        const uint32_t dst = abase + r * (KP * 2) + p * 16;
        asm volatile("global_load_async_to_lds_b128 %0, %1, off"
                     :: "v"(dst), "v"(src) : "memory");
    }

    // ---- edge features: f32 -> f16 conversion staged by VALU (overlaps async) ----
    if (tid < ROWS) {
        const int a = tid / MNBR, m = tid % MNBR;
        _Float16* dst = Ab + tid * KP;
        const float* ef = nbr_fea + ((atom0 + a) * MNBR + m) * NBF;
        #pragma unroll 8
        for (int j = 0; j < NBF; ++j) dst[128 + j] = (_Float16)ef[j];
        #pragma unroll
        for (int j = NBF; j < 64; ++j) dst[128 + j] = (_Float16)0.0f;
    }
    asm volatile("s_wait_asynccnt 0x0" ::: "memory");
    __syncthreads();

    const int wave = tid >> 5;
    const int lane = tid & 31;
    const int lrow = lane & 15;     // row (A/D M, B col N) within tile
    const int lhi  = lane >> 4;     // K half-group

    // 24 units = 6 row tiles x 4 (flt,core) column pairs; wave w owns cp = w,
    // B fragments are loop-invariant per wave and get hoisted to registers.
    for (int u = wave; u < 24; u += 4) {
        const int rt = u >> 2;      // row tile 0..5
        const int cp = u & 3;       // cols [cp*16) flt, [64+cp*16) core
        v8f acc0 = {};
        v8f acc1 = {};
        const _Float16* arow = Ab + (rt * 16 + lrow) * KP + lhi * 8;
        const _Float16* bp0  = wl + (cp * 16 + lrow) * KP + lhi * 8;
        const _Float16* bp1  = bp0 + 64 * KP;
        #pragma unroll
        for (int kc = 0; kc < 6; ++kc) {
            const int kb = kc * 32;
            V16U a, b0, b1;
            a.f[0]  = *reinterpret_cast<const float4*>(arow + kb);
            a.f[1]  = *reinterpret_cast<const float4*>(arow + kb + 16);
            b0.f[0] = *reinterpret_cast<const float4*>(bp0 + kb);
            b0.f[1] = *reinterpret_cast<const float4*>(bp0 + kb + 16);
            b1.f[0] = *reinterpret_cast<const float4*>(bp1 + kb);
            b1.f[1] = *reinterpret_cast<const float4*>(bp1 + kb + 16);
            acc0 = __builtin_amdgcn_wmma_f32_16x16x32_f16(false, a.h, false, b0.h,
                                                          (short)0, acc0, false, false);
            acc1 = __builtin_amdgcn_wmma_f32_16x16x32_f16(false, a.h, false, b1.h,
                                                          (short)0, acc1, false, false);
        }
        // epilogue: fold bias + BN1, gate, accumulate over neighbors
        const int cf = cp * 16 + lrow;    // flt column (0..63)
        const int cc = cf + 64;           // core column (64..127)
        const float sf = g1[cf] * BN_RS, tf = fmaf(cb[cf], sf, b1[cf]);
        const float sc = g1[cc] * BN_RS, tc = fmaf(cb[cc], sc, b1[cc]);
        #pragma unroll
        for (int g = 0; g < 8; ++g) {
            const int row = rt * 16 + g + 8 * lhi;        // local row 0..95
            const float fv = fmaf(acc0[g], sf, tf);
            const float cv = fmaf(acc1[g], sc, tc);
            const float prod = sigmoid_f(fv) * softplus_f(cv);
            atomicAdd(&nsum[(row / MNBR) * FDIM + cf], prod);
        }
    }
    __syncthreads();

    // ---- BN2 + residual softplus; write f32 and f16 mirror ----
    for (int i = tid; i < APB * FDIM; i += 128) {
        const int al = i >> 6, col = i & 63;
        const int ag = atom0 + al;
        const float ns = fmaf(nsum[i], g2[col] * BN_RS, b2[col]);
        const float v = softplus_f(x_in[ag * FDIM + col] + ns);
        x_out[ag * FDIM + col] = v;
        x16_out[ag * FDIM + col] = (_Float16)v;
    }
}

// ---------------- segment-sum pooling ----------------
__global__ void pool_kernel(const float* __restrict__ x, const int* __restrict__ seg,
                            float* __restrict__ csum, float* __restrict__ cnt) {
    int i = blockIdx.x * blockDim.x + threadIdx.x;
    if (i >= NATOM * FDIM) return;
    int a = i >> 6, col = i & 63;
    int s = seg[a];
    atomicAdd(&csum[s * FDIM + col], x[i]);
    if (col == 0) atomicAdd(&cnt[s], 1.0f);
}

// ---------------- per-crystal FC + output heads ----------------
__global__ __launch_bounds__(128) void head_kernel(
    const float* __restrict__ csum, const float* __restrict__ cnt,
    const float* __restrict__ extra,
    const float* __restrict__ W_fc, const float* __restrict__ b_fc,
    const float* __restrict__ W0, const float* __restrict__ bo0,
    const float* __restrict__ W1, const float* __restrict__ bo1,
    const float* __restrict__ W2, const float* __restrict__ bo2,
    float* __restrict__ out) {
    __shared__ float c[FDIM + EDIM];
    __shared__ float h[HDIM];
    __shared__ float z[4];
    const int b = blockIdx.x, tid = threadIdx.x;
    if (tid < FDIM) {
        c[tid] = csum[b * FDIM + tid] / fmaxf(cnt[b], 1.0f);
    } else if (tid < FDIM + EDIM) {
        c[tid] = extra[b * EDIM + (tid - FDIM)];
    }
    __syncthreads();
    float acc = b_fc[tid];
    #pragma unroll 4
    for (int k = 0; k < FDIM + EDIM; ++k) acc = fmaf(c[k], W_fc[k * HDIM + tid], acc);
    h[tid] = softplus_f(acc);
    __syncthreads();
    if (tid < 4) {
        float s = (tid == 0) ? bo0[0] : (tid == 3) ? bo2[0] : bo1[tid - 1];
        for (int j = 0; j < HDIM; ++j) {
            float w = (tid == 0) ? W0[j] : (tid == 3) ? W2[j] : W1[j * 2 + (tid - 1)];
            s = fmaf(h[j], w, s);
        }
        z[tid] = s;
    }
    __syncthreads();
    if (tid == 0) {
        out[b] = z[0];
        float m = fmaxf(z[1], z[2]);
        float lse = m + __logf(__expf(z[1] - m) + __expf(z[2] - m));
        out[NCRYS + 2 * b + 0] = z[1] - lse;
        out[NCRYS + 2 * b + 1] = z[2] - lse;
        out[NCRYS + 2 * NCRYS + b] = z[3];
    }
}

extern "C" void kernel_launch(void* const* d_in, const int* in_sizes, int n_in,
                              void* d_out, int out_size, void* d_ws, size_t ws_size,
                              hipStream_t stream) {
    const float* atom_fea   = (const float*)d_in[0];
    const float* nbr_fea    = (const float*)d_in[1];
    const int*   nbr_idx    = (const int*)d_in[2];
    const int*   seg_ids    = (const int*)d_in[3];
    const float* extra_fea  = (const float*)d_in[4];
    const float* W_embed    = (const float*)d_in[5];
    const float* b_embed    = (const float*)d_in[6];
    const float* conv_W     = (const float*)d_in[7];
    const float* conv_b     = (const float*)d_in[8];
    const float* bn1_g      = (const float*)d_in[9];
    const float* bn1_b      = (const float*)d_in[10];
    const float* bn2_g      = (const float*)d_in[11];
    const float* bn2_b      = (const float*)d_in[12];
    const float* W_extra    = (const float*)d_in[13];
    const float* b_extra    = (const float*)d_in[14];
    const float* bnx_g      = (const float*)d_in[15];
    const float* bnx_b      = (const float*)d_in[16];
    const float* W_fc       = (const float*)d_in[17];
    const float* b_fc       = (const float*)d_in[18];
    const float* W_out0     = (const float*)d_in[19];
    const float* b_out0     = (const float*)d_in[20];
    const float* W_out1     = (const float*)d_in[21];
    const float* b_out1     = (const float*)d_in[22];
    const float* W_out2     = (const float*)d_in[23];
    const float* b_out2     = (const float*)d_in[24];
    float* out = (float*)d_out;

    // workspace layout
    float* ws    = (float*)d_ws;
    float* x_a   = ws;                                   // 60000*64 f32
    float* x_b   = x_a + NATOM * FDIM;                   // 60000*64 f32
    float* csum  = x_b + NATOM * FDIM;                   // 2048*64
    float* cnt   = csum + NCRYS * FDIM;                  // 2048
    float* extra = cnt + NCRYS;                          // 2048*16
    _Float16* w16   = (_Float16*)(extra + NCRYS * EDIM); // 3*128*192 halves
    _Float16* x16_a = w16 + 3 * 2 * FDIM * KP;           // 60000*64 halves
    _Float16* x16_b = x16_a + NATOM * FDIM;              // 60000*64 halves

    // weight prep (f32 -> f16 transposed, K padded)
    {
        int n = 3 * 2 * FDIM * KP;
        prepw_kernel<<<(n + 255) / 256, 256, 0, stream>>>(conv_W, w16);
    }
    // atom embedding (writes f32 + f16 mirror)
    {
        int n = NATOM * FDIM;
        embed_kernel<<<(n + 255) / 256, 256, 0, stream>>>(atom_fea, W_embed, b_embed,
                                                          x_a, x16_a);
    }
    // extra embedding
    {
        int n = NCRYS * EDIM;
        extra_kernel<<<(n + 255) / 256, 256, 0, stream>>>(extra_fea, W_extra, b_extra,
                                                          bnx_g, bnx_b, extra);
    }
    // zero pooling scratch
    {
        int n = NCRYS * FDIM + NCRYS;
        zero_kernel<<<(n + 255) / 256, 256, 0, stream>>>(csum, n);
    }
    // 3 conv layers, ping-pong x / x16
    const float* xin = x_a;
    float* xout = x_b;
    const _Float16* x16in = x16_a;
    _Float16* x16out = x16_b;
    for (int l = 0; l < 3; ++l) {
        conv_kernel<<<NATOM / APB, 128, 0, stream>>>(
            xin, x16in, xout, x16out, nbr_fea, nbr_idx,
            w16 + l * 2 * FDIM * KP,
            conv_b + l * 2 * FDIM,
            bn1_g + l * 2 * FDIM, bn1_b + l * 2 * FDIM,
            bn2_g + l * FDIM, bn2_b + l * FDIM);
        const float* t = xout; xout = (float*)xin; xin = t;
        const _Float16* t16 = x16out; x16out = (_Float16*)x16in; x16in = t16;
    }
    // after 3 layers, xin points at the final x (x_b)
    {
        int n = NATOM * FDIM;
        pool_kernel<<<(n + 255) / 256, 256, 0, stream>>>(xin, seg_ids, csum, cnt);
    }
    // per-crystal FC + heads
    head_kernel<<<NCRYS, 128, 0, stream>>>(csum, cnt, extra, W_fc, b_fc,
                                           W_out0, b_out0, W_out1, b_out1,
                                           W_out2, b_out2, out);
}
